// VarRNNBase_1597727834468
// MI455X (gfx1250) — compile-verified, matrix operand order checked by reading, not run on hardware
//
#include <hip/hip_runtime.h>
#include <cstddef>
#include <cstdint>

// ---------------------------------------------------------------------------
// Types for CDNA5 WMMA (wave32, v_wmma_f32_16x16x32_f16)
// ---------------------------------------------------------------------------
typedef _Float16 v16h __attribute__((ext_vector_type(16)));
typedef _Float16 v8h  __attribute__((ext_vector_type(8)));
typedef float    v8f  __attribute__((ext_vector_type(8)));

union HFrag { v16h v; v8h h[2]; };

// Problem constants
constexpr int T_  = 512;
constexpr int B_  = 64;
constexpr int IN_ = 1024;
constexpr int H_  = 512;
constexpr int G4_ = 2048;           // 4*H
constexpr long M_ALL = (long)T_ * B_;   // 32768 rows for the big GEMM

// Workspace layout (bytes)
constexpr size_t OFF_XBF  = 0;                          // [T,B,1024] f16  (64 MB)
constexpr size_t OFF_WIHB = 67108864;                   // [4,2048,1024] f16 (16 MB)
constexpr size_t OFF_WHHB = 83886080;                   // [4,2048,512] f16  (8 MB)
constexpr size_t OFF_GX   = 92274688;                   // [2,32768,2048] f16 (256 MB)
constexpr size_t OFF_HBUF = 360710144;                  // [2 dir][2 parity][64][512] f16
constexpr size_t OFF_CTR  = 360972288;                  // 2 ints

// Scan LDS: W slice [128][520] halves + h [64][520] halves
constexpr int WL_STRIDE = 520;
constexpr int HL_STRIDE = 520;
constexpr int SCAN_LDS_BYTES = (128 * WL_STRIDE + 64 * HL_STRIDE) * 2; // 199680

// ---------------------------------------------------------------------------
// Activation helpers (use v_exp_f32 transcendental path)
// ---------------------------------------------------------------------------
__device__ __forceinline__ float sigmf(float x) {
  return 1.0f / (1.0f + __expf(-x));
}
__device__ __forceinline__ float tanh_fast(float x) {
  float e = __expf(-2.0f * fabsf(x));        // in (0,1], no overflow
  float t = (1.0f - e) / (1.0f + e);
  return x < 0.0f ? -t : t;
}

// ---------------------------------------------------------------------------
// fp32 -> fp16 conversion kernels
// ---------------------------------------------------------------------------
__global__ void cvt_f32_f16(const float* __restrict__ src,
                            _Float16* __restrict__ dst, long n) {
  long i = (long)blockIdx.x * blockDim.x + threadIdx.x;
  long stride = (long)gridDim.x * blockDim.x;
  for (; i < n; i += stride) dst[i] = (_Float16)src[i];
}

// x[T,B,1024] * mask_x[B,1024] -> f16
__global__ void cvt_x_mask(const float* __restrict__ x,
                           const float* __restrict__ mask_x,
                           _Float16* __restrict__ dst, long n) {
  long i = (long)blockIdx.x * blockDim.x + threadIdx.x;
  long stride = (long)gridDim.x * blockDim.x;
  for (; i < n; i += stride) {
    int feat = (int)(i & 1023);
    int b = (int)((i >> 10) & 63);
    dst[i] = (_Float16)(x[i] * mask_x[b * 1024 + feat]);
  }
}

__global__ void scan_init(_Float16* __restrict__ hbuf, int* __restrict__ ctr) {
  int i = blockIdx.x * blockDim.x + threadIdx.x;
  if (i < 2 * 2 * B_ * H_) hbuf[i] = (_Float16)0.0f;
  if (i < 2) ctr[i] = 0;
}

// ---------------------------------------------------------------------------
// Big time-parallel GEMM: Gx[dir][m][n] = sum_k X[m][k]*Wih[idx][n][k] + bias
// Block tile 128x128, K-chunk 32, 256 threads = 8 waves, 8 C-tiles per wave.
// ---------------------------------------------------------------------------
__global__ __launch_bounds__(256) void gates_gemm(
    const _Float16* __restrict__ X,      // [32768,1024] f16 (masked input)
    const _Float16* __restrict__ Wih,    // [4,2048,1024] f16
    const float* __restrict__ bih,       // [4,2048]
    const float* __restrict__ bhh,       // [4,2048]
    _Float16* __restrict__ Gx,           // [2,32768,2048] f16
    int layer)
{
  __shared__ _Float16 sa[128 * 40];
  __shared__ _Float16 sb[128 * 40];

  const int dir = blockIdx.z;
  const int idx = layer * 2 + dir;
  const _Float16* Wd = Wih + (size_t)idx * G4_ * IN_;
  const float* bi = bih + idx * G4_;
  const float* bh = bhh + idx * G4_;
  _Float16* G = Gx + (size_t)dir * M_ALL * G4_;

  const int m_base = blockIdx.y * 128;
  const int n_base = blockIdx.x * 128;
  const int tid = threadIdx.x;
  const int lane = tid & 31;
  const int w = tid >> 5;
  const int lo = lane & 15, hi = lane >> 4;
  const int mtb = (w >> 1) * 2;   // {0,2,4,6}
  const int ntb = (w & 1) * 4;    // {0,4}

  v8f acc[2][4];
#pragma unroll
  for (int i = 0; i < 2; ++i)
#pragma unroll
    for (int j = 0; j < 4; ++j)
#pragma unroll
      for (int r = 0; r < 8; ++r) acc[i][j][r] = 0.0f;

  const int srow = tid >> 1;
  const int spart = (tid & 1) * 16;

  for (int kc = 0; kc < IN_; kc += 32) {
    // cooperative staging: 128 rows x 32 halves for A and B
    const _Float16* ap = X + (size_t)(m_base + srow) * IN_ + kc + spart;
    *(v8h*)&sa[srow * 40 + spart]     = *(const v8h*)ap;
    *(v8h*)&sa[srow * 40 + spart + 8] = *(const v8h*)(ap + 8);
    const _Float16* bp = Wd + (size_t)(n_base + srow) * IN_ + kc + spart;
    *(v8h*)&sb[srow * 40 + spart]     = *(const v8h*)bp;
    *(v8h*)&sb[srow * 40 + spart + 8] = *(const v8h*)(bp + 8);
    if (kc + 32 < IN_) __builtin_prefetch(ap + 32, 0, 3);
    __syncthreads();

    HFrag af[2], bf[4];
    const int koff = hi * 8;
#pragma unroll
    for (int i = 0; i < 2; ++i) {
      int row = (mtb + i) * 16 + lo;
      af[i].h[0] = *(const v8h*)&sa[row * 40 + koff];
      af[i].h[1] = *(const v8h*)&sa[row * 40 + 16 + koff];
    }
#pragma unroll
    for (int j = 0; j < 4; ++j) {
      int row = (ntb + j) * 16 + lo;
      bf[j].h[0] = *(const v8h*)&sb[row * 40 + koff];
      bf[j].h[1] = *(const v8h*)&sb[row * 40 + 16 + koff];
    }
#pragma unroll
    for (int i = 0; i < 2; ++i)
#pragma unroll
      for (int j = 0; j < 4; ++j)
        acc[i][j] = __builtin_amdgcn_wmma_f32_16x16x32_f16(
            false, af[i].v, false, bf[j].v, (short)0, acc[i][j], false, false);
    __syncthreads();
  }

  // epilogue: + b_ih + b_hh, store f16
#pragma unroll
  for (int j = 0; j < 4; ++j) {
    int n = n_base + (ntb + j) * 16 + lo;
    float bias = bi[n] + bh[n];
#pragma unroll
    for (int i = 0; i < 2; ++i) {
#pragma unroll
      for (int r = 0; r < 8; ++r) {
        int m = m_base + (mtb + i) * 16 + r + 8 * hi;
        G[(size_t)m * G4_ + n] = (_Float16)(acc[i][j][r] + bias);
      }
    }
  }
}

// ---------------------------------------------------------------------------
// One gate-block accumulation for the recurrent GEMM (K=512).
// acc initialized from the precomputed x-gate preactivations.
// ---------------------------------------------------------------------------
__device__ __forceinline__ v8f gate_pass(const _Float16* __restrict__ Gt,
                                         const _Float16* __restrict__ hl,
                                         const _Float16* __restrict__ wl,
                                         int g, int mt, int u16, int u0,
                                         int lo, int hi)
{
  v8f a;
  const int n = g * H_ + u0 + u16 * 16 + lo;   // global gate column
#pragma unroll
  for (int r = 0; r < 8; ++r)
    a[r] = (float)Gt[(size_t)(mt * 16 + r + 8 * hi) * G4_ + n];

  const int nloc = g * 32 + u16 * 16 + lo;     // local W-slice row
  const int mrow = mt * 16 + lo;
  const int koff = hi * 8;
  const _Float16* arow = hl + mrow * HL_STRIDE;
  const _Float16* brow = wl + nloc * WL_STRIDE;
#pragma unroll 4
  for (int k0 = 0; k0 < H_; k0 += 32) {
    HFrag af, bf;
    af.h[0] = *(const v8h*)(arow + k0 + koff);
    af.h[1] = *(const v8h*)(arow + k0 + 16 + koff);
    bf.h[0] = *(const v8h*)(brow + k0 + koff);
    bf.h[1] = *(const v8h*)(brow + k0 + 16 + koff);
    a = __builtin_amdgcn_wmma_f32_16x16x32_f16(
        false, af.v, false, bf.v, (short)0, a, false, false);
  }
  return a;
}

// ---------------------------------------------------------------------------
// Sequential LSTM scan. grid = (16 unit-slices, 2 dirs), 256 threads (8 waves).
// Each WG owns 32 hidden units; W_hh slice [128 rows x 512] lives in LDS for
// the whole scan; h state (masked, f16, double-buffered by parity) in global,
// staged to LDS each step; steps synced via device-scope atomic barrier.
// wave w = (mt, u16): C tile rows mt*16..+15 (batch), units u16*16..+15.
// ---------------------------------------------------------------------------
__global__ __launch_bounds__(256) void lstm_scan(
    const _Float16* __restrict__ Gx,     // [2,32768,2048] f16
    const _Float16* __restrict__ Whh,    // [4,2048,512] f16
    _Float16* __restrict__ hbuf,         // [2 dir][2 parity][64][512] f16
    int* __restrict__ ctr,               // [2]
    const float* __restrict__ mask_h,    // [64,512]
    const float* __restrict__ mask_x,    // [64,1024]
    _Float16* __restrict__ outH,         // layer0: ys (masked) f16, else unused
    float* __restrict__ outF,            // layer1: ys fp32 into d_out
    float* __restrict__ hidOut,          // [4,64,512]
    float* __restrict__ cellOut,         // [4,64,512]
    int layer)
{
  extern __shared__ _Float16 lds[];
  _Float16* wl = lds;                       // [128][520]
  _Float16* hl = lds + 128 * WL_STRIDE;     // [64][520]

  const int dir = blockIdx.y;
  const int idx = layer * 2 + dir;
  const int u0  = blockIdx.x * 32;
  const int tid = threadIdx.x;
  const int lane = tid & 31;
  const int w = tid >> 5;
  const int lo = lane & 15, hi = lane >> 4;
  const int mt = w >> 1;       // 0..3 (batch 16-row tile)
  const int u16 = w & 1;       // 0..1 (unit 16-col tile)

  const _Float16* Gd = Gx + (size_t)dir * M_ALL * G4_;
  const _Float16* Wd = Whh + (size_t)idx * G4_ * H_;

  // ---- stage W_hh slice (rows n = g*512 + u0 + uu, g<4, uu<32) into LDS ----
  {
    int lr = tid >> 1;              // 0..127 local row
    int half = tid & 1;             // 256 halves each
    int g = lr >> 5, uu = lr & 31;
    int n = g * H_ + u0 + uu;
    const _Float16* src = Wd + (size_t)n * H_ + half * 256;
    _Float16* dst = wl + lr * WL_STRIDE + half * 256;
#pragma unroll
    for (int j = 0; j < 32; ++j)
      *(v8h*)(dst + j * 8) = *(const v8h*)(src + j * 8);
  }
  __syncthreads();

  const bool reverse = (dir == 1);
  const bool halfOut = (layer == 0);

  v8f c;
#pragma unroll
  for (int r = 0; r < 8; ++r) c[r] = 0.0f;

  for (int s = 0; s < T_; ++s) {
    const int t = reverse ? (T_ - 1 - s) : s;
    const _Float16* hbR = hbuf + ((size_t)dir * 2 + (s & 1)) * B_ * H_;
    _Float16*       hbW = hbuf + ((size_t)dir * 2 + ((s + 1) & 1)) * B_ * H_;

    // ---- stage h (masked, f16) into LDS: 64 rows x 512 halves ----
    {
      int row = tid >> 2, seg = tid & 3;   // 4 threads per row, 128 halves each
      const _Float16* src = hbR + (size_t)row * H_ + seg * 128;
      _Float16* dst = hl + row * HL_STRIDE + seg * 128;
#pragma unroll
      for (int j = 0; j < 16; ++j)
        *(v8h*)(dst + j * 8) = *(const v8h*)(src + j * 8);
    }
    __syncthreads();

    const _Float16* Gt = Gd + (size_t)t * B_ * G4_;

    // gates in order: g (tanh), i, f, o  — frees registers progressively
    v8f Gg = gate_pass(Gt, hl, wl, 2, mt, u16, u0, lo, hi);
#pragma unroll
    for (int r = 0; r < 8; ++r) Gg[r] = tanh_fast(Gg[r]);
    {
      v8f Gi = gate_pass(Gt, hl, wl, 0, mt, u16, u0, lo, hi);
#pragma unroll
      for (int r = 0; r < 8; ++r) Gg[r] *= sigmf(Gi[r]);   // p = sig(i)*tanh(g)
    }
    {
      v8f Gf = gate_pass(Gt, hl, wl, 1, mt, u16, u0, lo, hi);
#pragma unroll
      for (int r = 0; r < 8; ++r) c[r] = sigmf(Gf[r]) * c[r] + Gg[r];
    }
    v8f Go = gate_pass(Gt, hl, wl, 3, mt, u16, u0, lo, hi);

    // ---- elementwise outputs ----
#pragma unroll
    for (int r = 0; r < 8; ++r) {
      float hv = sigmf(Go[r]) * tanh_fast(c[r]);
      int b = mt * 16 + r + 8 * hi;
      int u = u0 + u16 * 16 + lo;
      hbW[(size_t)b * H_ + u] = (_Float16)(hv * mask_h[b * H_ + u]);
      int feat = dir * H_ + u;
      size_t oidx = (size_t)t * B_ * (2 * H_) + (size_t)b * (2 * H_) + feat;
      if (halfOut)
        outH[oidx] = (_Float16)(hv * mask_x[b * 1024 + feat]);
      else
        outF[oidx] = hv;
      if (s == T_ - 1) {
        hidOut[(size_t)idx * B_ * H_ + b * H_ + u]  = hv;
        cellOut[(size_t)idx * B_ * H_ + b * H_ + u] = c[r];
      }
    }

    // ---- device-scope step barrier (16 WGs per direction) ----
    __builtin_amdgcn_fence(__ATOMIC_RELEASE, "agent");
    __syncthreads();
    if (tid == 0) {
      __hip_atomic_fetch_add(&ctr[dir], 1, __ATOMIC_RELEASE,
                             __HIP_MEMORY_SCOPE_AGENT);
      int target = 16 * (s + 1);
      while (__hip_atomic_load(&ctr[dir], __ATOMIC_ACQUIRE,
                               __HIP_MEMORY_SCOPE_AGENT) < target)
        __builtin_amdgcn_s_sleep(1);
    }
    __syncthreads();
    __builtin_amdgcn_fence(__ATOMIC_ACQUIRE, "agent");
  }
}

// ---------------------------------------------------------------------------
// Host side
// ---------------------------------------------------------------------------
extern "C" void kernel_launch(void* const* d_in, const int* in_sizes, int n_in,
                              void* d_out, int out_size, void* d_ws, size_t ws_size,
                              hipStream_t stream) {
  const float* x      = (const float*)d_in[0];
  const float* Wih    = (const float*)d_in[1];
  const float* Whh    = (const float*)d_in[2];
  const float* bih    = (const float*)d_in[3];
  const float* bhh    = (const float*)d_in[4];
  const float* mask_x = (const float*)d_in[5];
  const float* mask_h = (const float*)d_in[6];
  float* out = (float*)d_out;

  char* ws = (char*)d_ws;
  _Float16* xbf  = (_Float16*)(ws + OFF_XBF);
  _Float16* wihb = (_Float16*)(ws + OFF_WIHB);
  _Float16* whhb = (_Float16*)(ws + OFF_WHHB);
  _Float16* Gx   = (_Float16*)(ws + OFF_GX);
  _Float16* hbuf = (_Float16*)(ws + OFF_HBUF);
  int*      ctr  = (int*)(ws + OFF_CTR);

  float* hidOut  = out + (size_t)T_ * B_ * (2 * H_);
  float* cellOut = hidOut + (size_t)4 * B_ * H_;

  hipFuncSetAttribute(reinterpret_cast<const void*>(lstm_scan),
                      hipFuncAttributeMaxDynamicSharedMemorySize,
                      SCAN_LDS_BYTES);

  // fp32 -> f16 conversions (mask_x folded into layer-0 input)
  cvt_x_mask<<<8192, 256, 0, stream>>>(x, mask_x, xbf, (long)T_ * B_ * IN_);
  cvt_f32_f16<<<8192, 256, 0, stream>>>(Wih, wihb, (long)4 * G4_ * IN_);
  cvt_f32_f16<<<4096, 256, 0, stream>>>(Whh, whhb, (long)4 * G4_ * H_);

  for (int layer = 0; layer < 2; ++layer) {
    gates_gemm<<<dim3(16, 256, 2), 256, 0, stream>>>(xbf, wihb, bih, bhh, Gx, layer);
    scan_init<<<(2 * 2 * B_ * H_ + 255) / 256, 256, 0, stream>>>(hbuf, ctr);
    lstm_scan<<<dim3(16, 2), 256, SCAN_LDS_BYTES, stream>>>(
        Gx, whhb, hbuf, ctr, mask_h, mask_x,
        (layer == 0) ? xbf : (_Float16*)nullptr,   // layer0 ys -> next-layer input (masked f16)
        (layer == 1) ? out : (float*)nullptr,      // layer1 ys -> d_out fp32
        hidOut, cellOut, layer);
  }
}
